// CausalSelfAttention_54331336294775
// MI455X (gfx1250) — matrix operand order repaired and checked by reading, not compiled, not run on hardware
//
#include <hip/hip_runtime.h>

// ---------------- problem constants ----------------
#define BB   2
#define TT   2048
#define DIM  1024
#define NH   16
#define KVH  4
#define HD   64
#define RANK 32
#define QMAXC 31.0f
#define SIGMAC 12.85f

typedef __attribute__((ext_vector_type(16))) _Float16 v16h;
typedef __attribute__((ext_vector_type(8)))  _Float16 v8h;
typedef __attribute__((ext_vector_type(8)))  float    v8f;
typedef __attribute__((ext_vector_type(4)))  float    vf4;
typedef __attribute__((ext_vector_type(4)))  unsigned int v4u;
typedef __attribute__((ext_vector_type(8)))  int      v8i;
typedef __attribute__((ext_vector_type(4)))  int      v4i;

// ---------------- WMMA helpers (gfx1250, wave32) ----------------
__device__ __forceinline__ v8f wmma16(const v16h& a, const v16h& b, const v8f& c) {
  // D = A(16x32 f16) x B(32x16 f16) + C(16x16 f32)
  return __builtin_amdgcn_wmma_f32_16x16x32_f16(false, a, false, b, (short)0, c, false, false);
}

// A fragment 16x32: element (m, kk) at base[m*lda + kk].
__device__ __forceinline__ v16h load_a(const _Float16* __restrict__ base, int lda, int lane) {
  int m  = lane & 15;
  int kb = (lane >> 4) * 8;
  const _Float16* p = base + (size_t)m * lda + kb;
  v8h lo = *(const v8h*)(p);        // K = kb .. kb+7
  v8h hi = *(const v8h*)(p + 16);   // K = kb+16 .. kb+23
  v16h r;
#pragma unroll
  for (int i = 0; i < 8; i++) { r[i] = lo[i]; r[i + 8] = hi[i]; }
  return r;
}

// B fragment 32x16 (K x N): element (kk, n) at base[n*nstride + kk]  (kk contiguous).
__device__ __forceinline__ v16h load_b_kcont(const _Float16* __restrict__ base, int nstride, int lane) {
  int n  = lane & 15;
  int kb = (lane >> 4) * 8;
  const _Float16* p = base + (size_t)n * nstride + kb;
  v8h lo = *(const v8h*)(p);
  v8h hi = *(const v8h*)(p + 16);
  v16h r;
#pragma unroll
  for (int i = 0; i < 8; i++) { r[i] = lo[i]; r[i + 8] = hi[i]; }
  return r;
}

// ---------------- Tensor Data Mover: 2D tile (f16 elements) global -> LDS ----------------
// D# per CDNA5 ISA ch.8: group0 = {flags, lds_addr, global_addr, type=2}, group1 = dims/strides.
__device__ __forceinline__ void tdm_load_2d(unsigned int lds_addr, const _Float16* gptr,
                                            unsigned int tdim0, unsigned int tdim1,
                                            unsigned int tile0, unsigned int tile1,
                                            unsigned long long stride0 /*elements*/) {
  unsigned long long ga = (unsigned long long)(size_t)gptr;   // byte address
  v4u g0;
  g0[0] = 1u;                                                  // count=1, user descriptor
  g0[1] = lds_addr;                                            // lds byte offset
  g0[2] = (unsigned int)(ga & 0xffffffffu);
  g0[3] = (unsigned int)((ga >> 32) & 0x01ffffffu) | 0x80000000u;  // addr[56:32] | type=2
  v8i g1;
  g1[0] = (int)(1u << 16);                                     // data_size = 2 bytes
  g1[1] = (int)((tdim0 & 0xffffu) << 16);                      // tensor_dim0 low16 (bits 63:48)
  g1[2] = (int)(((tdim0 >> 16) & 0xffffu) | ((tdim1 & 0xffffu) << 16));
  g1[3] = (int)(((tdim1 >> 16) & 0xffffu) | ((tile0 & 0xffffu) << 16));
  g1[4] = (int)(tile1 & 0xffffu);                              // tile_dim1, tile_dim2=0
  g1[5] = (int)(stride0 & 0xffffffffu);                        // tensor_dim0_stride low32
  g1[6] = (int)((stride0 >> 32) & 0xffffu);                    // stride0 hi16, stride1=0
  g1[7] = 0;
  v4i z4 = {0, 0, 0, 0};
  v8i z8 = {0, 0, 0, 0, 0, 0, 0, 0};
  __builtin_amdgcn_tensor_load_to_lds(g0, g1, z4, z4, z8, 0);  // 6-arg (clang-23) form
}

// ---------------- block reductions (128 threads = 4 wave32) ----------------
__device__ __forceinline__ float block_sum(float v, float* red) {
#pragma unroll
  for (int off = 16; off >= 1; off >>= 1) v += __shfl_xor(v, off);
  __syncthreads();
  if ((threadIdx.x & 31) == 0) red[threadIdx.x >> 5] = v;
  __syncthreads();
  return red[0] + red[1] + red[2] + red[3];
}
__device__ __forceinline__ float block_max(float v, float* red) {
#pragma unroll
  for (int off = 16; off >= 1; off >>= 1) v = fmaxf(v, __shfl_xor(v, off));
  __syncthreads();
  if ((threadIdx.x & 31) == 0) red[threadIdx.x >> 5] = v;
  __syncthreads();
  return fmaxf(fmaxf(red[0], red[1]), fmaxf(red[2], red[3]));
}

// ---------------- int6 row-wise fake-quant -> f16 dequantized weights ----------------
__global__ void fakequant_kernel(const float* __restrict__ W, _Float16* __restrict__ Wout) {
  __shared__ float red[4];
  const int K = DIM;
  int r   = blockIdx.x;
  int tid = threadIdx.x;
  const float* row = W + (size_t)r * K;
  float vals[8];
  float sum = 0.f, sumsq = 0.f, amax = 0.f;
#pragma unroll
  for (int i = 0; i < 8; i++) {
    float v = row[i * 128 + tid];
    vals[i] = v;
    sum += v; sumsq += v * v; amax = fmaxf(amax, fabsf(v));
  }
  sum   = block_sum(sum, red);
  sumsq = block_sum(sumsq, red);
  amax  = block_max(amax, red);
  float mean   = sum / (float)K;
  float var    = fmaxf(sumsq / (float)K - mean * mean, 0.f);
  float rstd   = fmaxf(sqrtf(var), 1e-8f);
  float absmax = fmaxf(amax, 1.0f / QMAXC);
  float s1 = fmaxf(fmaxf(rstd * SIGMAC, absmax / QMAXC) / QMAXC, 1.0f / QMAXC);
  float s2 = fmaxf(absmax / QMAXC, 1.0f / QMAXC);
  float e1 = 0.f, e2 = 0.f;
#pragma unroll
  for (int i = 0; i < 8; i++) {
    float t  = vals[i];
    float q1 = fminf(fmaxf(rintf(t / s1), -QMAXC), QMAXC);
    float q2 = fminf(fmaxf(rintf(t / s2), -QMAXC), QMAXC);
    float d1 = q1 * s1 - t, d2 = q2 * s2 - t;
    e1 += d1 * d1; e2 += d2 * d2;
  }
  e1 = block_sum(e1, red);
  e2 = block_sum(e2, red);
  float ssel = (e1 < e2) ? s1 : s2;
  float sh   = (float)(_Float16)ssel;   // f16 round-trip of the scale (as in reference)
#pragma unroll
  for (int i = 0; i < 8; i++) {
    float q = fminf(fmaxf(rintf(vals[i] / ssel), -QMAXC), QMAXC);
    Wout[(size_t)r * K + i * 128 + tid] = (_Float16)(q * sh);
  }
}

// ---------------- elementwise converts ----------------
__global__ void conv_f16_kernel(const float* __restrict__ src, _Float16* __restrict__ dst, int n) {
  int i = blockIdx.x * blockDim.x + threadIdx.x;
  if (i < n) dst[i] = (_Float16)src[i];
}
__global__ void tconv_kernel(const float* __restrict__ src, _Float16* __restrict__ dst, int R, int C) {
  int i = blockIdx.x * blockDim.x + threadIdx.x;
  if (i >= R * C) return;
  int r = i / C, c = i - r * C;
  dst[(size_t)c * R + r] = (_Float16)src[i];
}

// ---------------- templated WMMA GEMM:  out = X @ W^T  (+ T1 @ B2t^T) ----------------
template <int MT, int NT, bool LORA, bool OUTF32>
__global__ __launch_bounds__(128) void gemm_kernel(
    const _Float16* __restrict__ X, const _Float16* __restrict__ W,
    const _Float16* __restrict__ T1, const _Float16* __restrict__ B2t,
    _Float16* __restrict__ outh, float* __restrict__ outf,
    int M, int N, int K) {
  (void)M;
  int lane = threadIdx.x & 31;
  int wave = threadIdx.x >> 5;
  int m0 = blockIdx.x * (4 * MT * 16) + wave * (MT * 16);
  int n0 = blockIdx.y * (NT * 16);
  v8f c[MT][NT];
#pragma unroll
  for (int mi = 0; mi < MT; mi++)
#pragma unroll
    for (int ni = 0; ni < NT; ni++)
#pragma unroll
      for (int v = 0; v < 8; v++) c[mi][ni][v] = 0.f;

  for (int k0 = 0; k0 < K; k0 += 32) {
    if (k0 + 32 < K) {
      __builtin_prefetch(X + (size_t)m0 * K + k0 + 32, 0, 3);
      __builtin_prefetch(W + (size_t)n0 * K + k0 + 32, 0, 3);
    }
    v16h a[MT], b[NT];
#pragma unroll
    for (int mi = 0; mi < MT; mi++)
      a[mi] = load_a(X + (size_t)(m0 + mi * 16) * K + k0, K, lane);
#pragma unroll
    for (int ni = 0; ni < NT; ni++)
      b[ni] = load_b_kcont(W + (size_t)(n0 + ni * 16) * K + k0, K, lane);
#pragma unroll
    for (int mi = 0; mi < MT; mi++)
#pragma unroll
      for (int ni = 0; ni < NT; ni++)
        c[mi][ni] = wmma16(a[mi], b[ni], c[mi][ni]);
  }
  if (LORA) {  // one extra K-slab of 32 = RANK
    v16h a[MT], b[NT];
#pragma unroll
    for (int mi = 0; mi < MT; mi++)
      a[mi] = load_a(T1 + (size_t)(m0 + mi * 16) * RANK, RANK, lane);
#pragma unroll
    for (int ni = 0; ni < NT; ni++)
      b[ni] = load_b_kcont(B2t + (size_t)(n0 + ni * 16) * RANK, RANK, lane);
#pragma unroll
    for (int mi = 0; mi < MT; mi++)
#pragma unroll
      for (int ni = 0; ni < NT; ni++)
        c[mi][ni] = wmma16(a[mi], b[ni], c[mi][ni]);
  }
  int nc = lane & 15, mh = (lane >> 4) * 8;
#pragma unroll
  for (int mi = 0; mi < MT; mi++)
#pragma unroll
    for (int ni = 0; ni < NT; ni++)
#pragma unroll
      for (int v = 0; v < 8; v++) {
        size_t idx = (size_t)(m0 + mi * 16 + mh + v) * N + (n0 + ni * 16 + nc);
        if (OUTF32) outf[idx] = c[mi][ni][v];
        else        outh[idx] = (_Float16)c[mi][ni][v];
      }
}

// ---------------- RoPE + head re-layout ----------------
#define LOG2_ROPE 13.287712379549449f   // log2(10000)
__global__ void rope_q_kernel(const _Float16* __restrict__ q16, const float* __restrict__ gain,
                              _Float16* __restrict__ qh) {
  int idx = blockIdx.x * blockDim.x + threadIdx.x;
  int j = idx & 31;
  int t = (idx >> 5) & (TT - 1);
  int h = (idx >> 16) & (NH - 1);
  int b = idx >> 20;
  float inv = exp2f(-(float)(2 * j) * (LOG2_ROPE / (float)HD));
  float sn, cs; sincosf((float)t * inv, &sn, &cs);
  size_t src = ((size_t)(b * TT + t)) * DIM + h * HD + j;
  float x1 = (float)q16[src], x2 = (float)q16[src + 32];
  float g = gain[h];
  size_t dst = (((size_t)(b * NH + h)) * TT + t) * HD + j;
  qh[dst]      = (_Float16)((x1 * cs + x2 * sn) * g);
  qh[dst + 32] = (_Float16)((-x1 * sn + x2 * cs) * g);
}
__global__ void rope_k_kernel(const _Float16* __restrict__ k16, _Float16* __restrict__ kh) {
  int idx = blockIdx.x * blockDim.x + threadIdx.x;
  int j = idx & 31;
  int t = (idx >> 5) & (TT - 1);
  int kv = (idx >> 16) & (KVH - 1);
  int b = idx >> 18;
  float inv = exp2f(-(float)(2 * j) * (LOG2_ROPE / (float)HD));
  float sn, cs; sincosf((float)t * inv, &sn, &cs);
  size_t src = ((size_t)(b * TT + t)) * (KVH * HD) + kv * HD + j;
  float x1 = (float)k16[src], x2 = (float)k16[src + 32];
  size_t dst = (((size_t)(b * KVH + kv)) * TT + t) * HD + j;
  kh[dst]      = (_Float16)(x1 * cs + x2 * sn);
  kh[dst + 32] = (_Float16)(-x1 * sn + x2 * cs);
}
// v16 [B*T, KVH*HD] -> vt [B][KVH][HD][T]
__global__ void vtrans_kernel(const _Float16* __restrict__ v16, _Float16* __restrict__ vt) {
  int idx = blockIdx.x * blockDim.x + threadIdx.x;
  int d  = idx & 63;
  int kv = (idx >> 6) & (KVH - 1);
  int t  = (idx >> 8) & (TT - 1);
  int b  = idx >> 19;
  vt[(((size_t)(b * KVH + kv)) * HD + d) * TT + t] = v16[idx];
}

// ---------------- flash attention (causal, GQA) ----------------
// grid: (TT/128, BB*NH); block 128 = 4 waves; each wave: 32 query rows x HD=64.
// Wave 0 streams shared K/V chunks into LDS with the TDM (double-buffered);
// softmax is row-per-lane through a transposed LDS score tile (no bpermutes).
__global__ __launch_bounds__(128) void attn_kernel(
    const _Float16* __restrict__ qh, const _Float16* __restrict__ kh,
    const _Float16* __restrict__ vt, const float* __restrict__ head_gate,
    _Float16* __restrict__ y16) {
  __shared__ _Float16 kbuf[2][32][64];   // [buf][key][d]
  __shared__ _Float16 vbuf[2][64][32];   // [buf][d][key]
  __shared__ float    sbufT[4][32][36];  // per-wave scores, [col][row] (pitch 36)
  __shared__ _Float16 pbuf[4][32][40];   // per-wave P tile, [row][col] (pitch 40)
  __shared__ float    scbuf[4][32];      // per-wave per-row rescale broadcast

  int lane = threadIdx.x & 31;
  int wave = threadIdx.x >> 5;
  int bh = blockIdx.y;
  int b  = bh >> 4;
  int h  = bh & (NH - 1);
  int kv = h >> 2;
  int q0blk = blockIdx.x * 128;
  int q0 = q0blk + wave * 32;

  const _Float16* qbase = qh + (((size_t)(b * NH + h)) * TT + q0) * HD;
  const _Float16* kbase = kh + ((size_t)(b * KVH + kv)) * TT * HD;
  const _Float16* vbase = vt + ((size_t)(b * KVH + kv)) * HD * TT;

  // Q fragments, pre-scaled by 1/sqrt(HD) = 0.125 (exact in f16)
  v16h qa[2][2];
#pragma unroll
  for (int mt = 0; mt < 2; mt++)
#pragma unroll
    for (int sl = 0; sl < 2; sl++) {
      qa[mt][sl] = load_a(qbase + (size_t)(mt * 16) * HD + sl * 32, HD, lane);
#pragma unroll
      for (int i = 0; i < 16; i++) qa[mt][sl][i] = qa[mt][sl][i] * (_Float16)0.125f;
    }

  v8f o[2][4];
#pragma unroll
  for (int mt = 0; mt < 2; mt++)
#pragma unroll
    for (int dt = 0; dt < 4; dt++)
#pragma unroll
      for (int v = 0; v < 8; v++) o[mt][dt][v] = 0.f;
  float mst = -3.0e38f, lst = 0.f;   // this lane owns local query row = lane

  int mh = (lane >> 4) * 8, nc = lane & 15;
  int nch  = (q0blk >> 5) + 4;   // block-uniform #chunks (last wave's causal extent)
  int mych = (q0 >> 5) + 1;      // chunks this wave actually computes

  if (wave == 0) {  // stage chunk 0
    tdm_load_2d((unsigned int)(size_t)&kbuf[0][0][0], kbase, HD, TT, HD, 32, HD);
    tdm_load_2d((unsigned int)(size_t)&vbuf[0][0][0], vbase, TT, HD, 32, HD, TT);
  }

  for (int ci = 0; ci < nch; ci++) {
    int cur = ci & 1;
    if (wave == 0) {
      if (ci + 1 < nch) {       // prefetch next chunk into the other buffer
        int n1 = (ci + 1) << 5;
        tdm_load_2d((unsigned int)(size_t)&kbuf[cur ^ 1][0][0], kbase + (size_t)n1 * HD, HD, TT, HD, 32, HD);
        tdm_load_2d((unsigned int)(size_t)&vbuf[cur ^ 1][0][0], vbase + n1, TT, HD, 32, HD, TT);
        __builtin_amdgcn_s_wait_tensorcnt(2);   // chunk ci's two loads have landed
      } else {
        __builtin_amdgcn_s_wait_tensorcnt(0);
      }
    }
    __syncthreads();            // K/V chunk ci visible to all waves

    if (ci < mych) {
      int n0 = ci << 5;
      // S = Q K^T from LDS K tile
      v16h kb[2][2];
#pragma unroll
      for (int sl = 0; sl < 2; sl++)
#pragma unroll
        for (int nt = 0; nt < 2; nt++)
          kb[sl][nt] = load_b_kcont(&kbuf[cur][nt * 16][sl * 32], HD, lane);
      v8f s[2][2];
#pragma unroll
      for (int mt = 0; mt < 2; mt++)
#pragma unroll
        for (int nt = 0; nt < 2; nt++) {
          v8f z;
#pragma unroll
          for (int v = 0; v < 8; v++) z[v] = 0.f;
          z = wmma16(qa[mt][0], kb[0][nt], z);
          s[mt][nt] = wmma16(qa[mt][1], kb[1][nt], z);
        }
      // store S transposed (rows contiguous per lane -> 2x b128 per frag)
#pragma unroll
      for (int mt = 0; mt < 2; mt++)
#pragma unroll
        for (int nt = 0; nt < 2; nt++) {
          float* basep = &sbufT[wave][nt * 16 + nc][mt * 16 + mh];
          vf4 lo = {s[mt][nt][0], s[mt][nt][1], s[mt][nt][2], s[mt][nt][3]};
          vf4 hi = {s[mt][nt][4], s[mt][nt][5], s[mt][nt][6], s[mt][nt][7]};
          *(vf4*)(basep)     = lo;
          *(vf4*)(basep + 4) = hi;
        }
      asm volatile("s_wait_dscnt 0" ::: "memory");

      // row-per-lane online softmax (causal mask via column limit)
      int climit = q0 + lane - n0;     // valid cols: c <= climit
      float sv[32];
#pragma unroll
      for (int c = 0; c < 32; c++) {
        float t = sbufT[wave][c][lane];
        sv[c] = (c <= climit) ? t : -1.0e30f;
      }
      float mx = sv[0];
#pragma unroll
      for (int c = 1; c < 32; c++) mx = fmaxf(mx, sv[c]);
      float mnew = fmaxf(mst, mx);
      float sc   = __expf(mst - mnew);
      mst = mnew;
      float rs = 0.f;
      v8h pk[4];
#pragma unroll
      for (int c = 0; c < 32; c++) {
        float p = __expf(sv[c] - mnew);
        rs += p;
        pk[c >> 3][c & 7] = (_Float16)p;
      }
      lst = lst * sc + rs;
      scbuf[wave][lane] = sc;
#pragma unroll
      for (int cc = 0; cc < 4; cc++)
        *(v8h*)&pbuf[wave][lane][cc * 8] = pk[cc];
      asm volatile("s_wait_dscnt 0" ::: "memory");

      // rescale O by per-row sc (LDS broadcast), then O += P V
#pragma unroll
      for (int mt = 0; mt < 2; mt++) {
        vf4 s0 = *(const vf4*)&scbuf[wave][mt * 16 + mh];
        vf4 s1 = *(const vf4*)&scbuf[wave][mt * 16 + mh + 4];
#pragma unroll
        for (int dt = 0; dt < 4; dt++)
#pragma unroll
          for (int v = 0; v < 8; v++)
            o[mt][dt][v] *= (v < 4) ? s0[v] : s1[v - 4];
      }
      v16h pa[2], vb[4];
#pragma unroll
      for (int mt = 0; mt < 2; mt++)
        pa[mt] = load_a(&pbuf[wave][mt * 16][0], 40, lane);
#pragma unroll
      for (int dt = 0; dt < 4; dt++)
        vb[dt] = load_b_kcont(&vbuf[cur][dt * 16][0], 32, lane);
#pragma unroll
      for (int mt = 0; mt < 2; mt++)
#pragma unroll
        for (int dt = 0; dt < 4; dt++)
          o[mt][dt] = wmma16(pa[mt], vb[dt], o[mt][dt]);
    }
    asm volatile("s_wait_dscnt 0" ::: "memory");
    __syncthreads();            // everyone done reading buf[cur] before TDM reuses it
  }

  // epilogue: gate * O / l  (per-row l broadcast through LDS)
  float gate = head_gate[h];
  scbuf[wave][lane] = gate / lst;
  asm volatile("s_wait_dscnt 0" ::: "memory");
#pragma unroll
  for (int mt = 0; mt < 2; mt++) {
    vf4 l0 = *(const vf4*)&scbuf[wave][mt * 16 + mh];
    vf4 l1 = *(const vf4*)&scbuf[wave][mt * 16 + mh + 4];
#pragma unroll
    for (int v = 0; v < 8; v++) {
      float rl = (v < 4) ? l0[v] : l1[v - 4];
      int mrow = q0 + mt * 16 + mh + v;
#pragma unroll
      for (int dt = 0; dt < 4; dt++) {
        size_t dst = ((size_t)(b * TT + mrow)) * DIM + h * HD + dt * 16 + nc;
        y16[dst] = (_Float16)(o[mt][dt][v] * rl);
      }
    }
  }
}

// ---------------- host-side orchestration ----------------
extern "C" void kernel_launch(void* const* d_in, const int* in_sizes, int n_in,
                              void* d_out, int out_size, void* d_ws, size_t ws_size,
                              hipStream_t stream) {
  (void)in_sizes; (void)n_in; (void)out_size;
  const float* x    = (const float*)d_in[0];
  const float* Wq   = (const float*)d_in[1];
  const float* Wk   = (const float*)d_in[2];
  const float* Wv   = (const float*)d_in[3];
  const float* Wo   = (const float*)d_in[4];
  const float* qA   = (const float*)d_in[5];
  const float* qB   = (const float*)d_in[6];
  const float* kA   = (const float*)d_in[7];
  const float* kB   = (const float*)d_in[8];
  const float* vA   = (const float*)d_in[9];
  const float* vB   = (const float*)d_in[10];
  const float* pA   = (const float*)d_in[11];
  const float* pB   = (const float*)d_in[12];
  const float* qg   = (const float*)d_in[13];
  const float* hg   = (const float*)d_in[14];

  const int M = BB * TT;  // 4096
  char* ws = (char*)d_ws;
  size_t off = 0;
  auto alloc = [&](size_t bytes) -> char* {
    char* p = ws + off;
    off += (bytes + 255) & ~(size_t)255;
    return p;
  };
  _Float16* Wq16 = (_Float16*)alloc((size_t)DIM * DIM * 2);
  _Float16* Wk16 = (_Float16*)alloc((size_t)KVH * HD * DIM * 2);
  _Float16* Wv16 = (_Float16*)alloc((size_t)KVH * HD * DIM * 2);
  _Float16* Wo16 = (_Float16*)alloc((size_t)DIM * DIM * 2);
  _Float16* x16  = (_Float16*)alloc((size_t)M * DIM * 2);
  _Float16* qAt  = (_Float16*)alloc((size_t)RANK * DIM * 2);
  _Float16* kAt  = (_Float16*)alloc((size_t)RANK * DIM * 2);
  _Float16* vAt  = (_Float16*)alloc((size_t)RANK * DIM * 2);
  _Float16* pAt  = (_Float16*)alloc((size_t)RANK * DIM * 2);
  _Float16* qBt  = (_Float16*)alloc((size_t)DIM * RANK * 2);
  _Float16* kBt  = (_Float16*)alloc((size_t)KVH * HD * RANK * 2);
  _Float16* vBt  = (_Float16*)alloc((size_t)KVH * HD * RANK * 2);
  _Float16* pBt  = (_Float16*)alloc((size_t)DIM * RANK * 2);
  _Float16* tq   = (_Float16*)alloc((size_t)M * RANK * 2);
  _Float16* tk   = (_Float16*)alloc((size_t)M * RANK * 2);
  _Float16* tv   = (_Float16*)alloc((size_t)M * RANK * 2);
  _Float16* tp   = (_Float16*)alloc((size_t)M * RANK * 2);
  _Float16* q16  = (_Float16*)alloc((size_t)M * DIM * 2);
  _Float16* k16  = (_Float16*)alloc((size_t)M * KVH * HD * 2);
  _Float16* v16  = (_Float16*)alloc((size_t)M * KVH * HD * 2);
  _Float16* qhb  = (_Float16*)alloc((size_t)BB * NH * TT * HD * 2);
  _Float16* khb  = (_Float16*)alloc((size_t)BB * KVH * TT * HD * 2);
  _Float16* vtb  = (_Float16*)alloc((size_t)BB * KVH * HD * TT * 2);
  _Float16* y16  = (_Float16*)alloc((size_t)M * DIM * 2);
  if (off > ws_size) return;

  // 1) fake-quant dequantized weights (f16)
  fakequant_kernel<<<DIM, 128, 0, stream>>>(Wq, Wq16);
  fakequant_kernel<<<KVH * HD, 128, 0, stream>>>(Wk, Wk16);
  fakequant_kernel<<<KVH * HD, 128, 0, stream>>>(Wv, Wv16);
  fakequant_kernel<<<DIM, 128, 0, stream>>>(Wo, Wo16);

  // 2) converts / transposes
  conv_f16_kernel<<<(M * DIM) / 256, 256, 0, stream>>>(x, x16, M * DIM);
  tconv_kernel<<<(DIM * RANK + 255) / 256, 256, 0, stream>>>(qA, qAt, DIM, RANK);
  tconv_kernel<<<(DIM * RANK + 255) / 256, 256, 0, stream>>>(kA, kAt, DIM, RANK);
  tconv_kernel<<<(DIM * RANK + 255) / 256, 256, 0, stream>>>(vA, vAt, DIM, RANK);
  tconv_kernel<<<(DIM * RANK + 255) / 256, 256, 0, stream>>>(pA, pAt, DIM, RANK);
  tconv_kernel<<<(RANK * DIM + 255) / 256, 256, 0, stream>>>(qB, qBt, RANK, DIM);
  tconv_kernel<<<(RANK * KVH * HD + 255) / 256, 256, 0, stream>>>(kB, kBt, RANK, KVH * HD);
  tconv_kernel<<<(RANK * KVH * HD + 255) / 256, 256, 0, stream>>>(vB, vBt, RANK, KVH * HD);
  tconv_kernel<<<(RANK * DIM + 255) / 256, 256, 0, stream>>>(pB, pBt, RANK, DIM);

  // 3) LoRA left factors: t = x @ A
  gemm_kernel<2, 2, false, false><<<dim3(M / 128, 1), 128, 0, stream>>>(
      x16, qAt, nullptr, nullptr, tq, nullptr, M, RANK, DIM);
  gemm_kernel<2, 2, false, false><<<dim3(M / 128, 1), 128, 0, stream>>>(
      x16, kAt, nullptr, nullptr, tk, nullptr, M, RANK, DIM);
  gemm_kernel<2, 2, false, false><<<dim3(M / 128, 1), 128, 0, stream>>>(
      x16, vAt, nullptr, nullptr, tv, nullptr, M, RANK, DIM);

  // 4) projections with fused LoRA
  gemm_kernel<4, 4, true, false><<<dim3(M / 256, DIM / 64), 128, 0, stream>>>(
      x16, Wq16, tq, qBt, q16, nullptr, M, DIM, DIM);
  gemm_kernel<4, 4, true, false><<<dim3(M / 256, (KVH * HD) / 64), 128, 0, stream>>>(
      x16, Wk16, tk, kBt, k16, nullptr, M, KVH * HD, DIM);
  gemm_kernel<4, 4, true, false><<<dim3(M / 256, (KVH * HD) / 64), 128, 0, stream>>>(
      x16, Wv16, tv, vBt, v16, nullptr, M, KVH * HD, DIM);

  // 5) RoPE + re-layout
  rope_q_kernel<<<(BB * NH * TT * (HD / 2)) / 256, 256, 0, stream>>>(q16, qg, qhb);
  rope_k_kernel<<<(BB * KVH * TT * (HD / 2)) / 256, 256, 0, stream>>>(k16, khb);
  vtrans_kernel<<<(BB * TT * KVH * HD) / 256, 256, 0, stream>>>(v16, vtb);

  // 6) causal flash attention (TDM-staged K/V, WMMA, head gates)
  attn_kernel<<<dim3(TT / 128, BB * NH), 128, 0, stream>>>(qhb, khb, vtb, hg, y16);

  // 7) output projection with fused LoRA, f32 out
  gemm_kernel<2, 2, false, false><<<dim3(M / 128, 1), 128, 0, stream>>>(
      y16, pAt, nullptr, nullptr, tp, nullptr, M, RANK, DIM);
  gemm_kernel<4, 4, true, true><<<dim3(M / 256, DIM / 64), 128, 0, stream>>>(
      y16, Wo16, tp, pBt, nullptr, (float*)d_out, M, DIM, DIM);
}
